// GCN_22436909154907
// MI455X (gfx1250) — matrix operand order repaired and checked by reading, not compiled, not run on hardware
//
#include <hip/hip_runtime.h>
#include <stdint.h>
#include <stddef.h>

#define NNODES 50000
#define NEDGES 800000
#define FIN    512
#define FHID   128
#define FOUT   20

typedef __attribute__((ext_vector_type(16))) __bf16 v16bf;
typedef __attribute__((ext_vector_type(8)))  float  v8f;

union Frag16 {
  v16bf v;
  uint4 q[2];
  unsigned int d[8];
  unsigned short u[16];
};

__device__ __forceinline__ unsigned short f32_to_bf16(float f) {
  unsigned int b = __float_as_uint(f);
  b += 0x7FFFu + ((b >> 16) & 1u);   // round-to-nearest-even
  return (unsigned short)(b >> 16);
}

__device__ __forceinline__ unsigned int pack_bf16x2(float a, float b) {
  return (unsigned int)f32_to_bf16(a) | ((unsigned int)f32_to_bf16(b) << 16);
}

// async DMA: 16 bytes global -> LDS, tracked by ASYNCcnt (CDNA5)
__device__ __forceinline__ void async_cp16(unsigned lds_off, const void* gaddr) {
  asm volatile("global_load_async_to_lds_b128 %0, %1, off"
               :: "v"(lds_off), "v"(gaddr) : "memory");
}
__device__ __forceinline__ void wait_async0() {
  asm volatile("s_wait_asynccnt 0x0" ::: "memory");
}
__device__ __forceinline__ unsigned lds_off_of(const void* p) {
  return (unsigned)(uintptr_t)p;   // generic LDS addr: low 32 bits = LDS offset
}

// ---------------------------------------------------------------- small utils
__global__ void k_fill(float* __restrict__ p, float v, int n) {
  int i = blockIdx.x * blockDim.x + threadIdx.x;
  if (i < n) p[i] = v;
}

__global__ void k_init_bias(float* __restrict__ p, const float* __restrict__ b,
                            int feat, int n) {
  int i = blockIdx.x * blockDim.x + threadIdx.x;
  if (i < n) p[i] = b[i % feat];
}

__global__ void k_deg_accum(const long long* __restrict__ ei,
                            const float* __restrict__ ew,
                            float* __restrict__ deg) {
  int e = blockIdx.x * blockDim.x + threadIdx.x;
  if (e < NEDGES) {
    int c = (int)ei[NEDGES + e];
    atomicAdd(&deg[c], ew[e]);
  }
}

__global__ void k_dinv(const float* __restrict__ deg, float* __restrict__ dinv) {
  int i = blockIdx.x * blockDim.x + threadIdx.x;
  if (i < NNODES) {
    float d = deg[i];
    dinv[i] = (d > 0.0f) ? rsqrtf(d) : 0.0f;
  }
}

__global__ void k_norm(const long long* __restrict__ ei,
                       const float* __restrict__ ew,
                       const float* __restrict__ dinv,
                       float* __restrict__ norm) {
  int e = blockIdx.x * blockDim.x + threadIdx.x;
  if (e < NEDGES) {
    int r = (int)ei[e];
    int c = (int)ei[NEDGES + e];
    norm[e] = dinv[r] * ew[e] * dinv[c];
  }
}

// f32 -> bf16 (RNE), 4 elems/thread; n must be a multiple of 4
__global__ void k_cvt_bf16(const float* __restrict__ src,
                           unsigned short* __restrict__ dst, int n4) {
  int i = blockIdx.x * blockDim.x + threadIdx.x;
  if (i < n4) {
    const float4 v = *(const float4*)(src + (size_t)i * 4);
    uint2 p; p.x = pack_bf16x2(v.x, v.y); p.y = pack_bf16x2(v.z, v.w);
    *(uint2*)(dst + (size_t)i * 4) = p;
  }
}

// relu + f32 -> bf16, 4 elems/thread
__global__ void k_relu_cvt(const float* __restrict__ src,
                           unsigned short* __restrict__ dst, int n4) {
  int i = blockIdx.x * blockDim.x + threadIdx.x;
  if (i < n4) {
    float4 v = *(const float4*)(src + (size_t)i * 4);
    v.x = fmaxf(v.x, 0.f); v.y = fmaxf(v.y, 0.f);
    v.z = fmaxf(v.z, 0.f); v.w = fmaxf(v.w, 0.f);
    uint2 p; p.x = pack_bf16x2(v.x, v.y); p.y = pack_bf16x2(v.z, v.w);
    *(uint2*)(dst + (size_t)i * 4) = p;
  }
}

// W1[512,128] f32 -> W1T[128,512] bf16
__global__ void k_w1t(const float* __restrict__ W1, unsigned short* __restrict__ T) {
  int i = blockIdx.x * blockDim.x + threadIdx.x;
  if (i < FHID * FIN) {
    int n = i >> 9, k = i & 511;
    T[i] = f32_to_bf16(W1[(size_t)k * FHID + n]);
  }
}

// W2[128,20] f32 -> W2T[32,128] bf16 zero-padded
__global__ void k_w2t(const float* __restrict__ W2, unsigned short* __restrict__ T) {
  int i = blockIdx.x * blockDim.x + threadIdx.x;
  if (i < 32 * FHID) {
    int n = i >> 7, k = i & 127;
    T[i] = (n < FOUT) ? f32_to_bf16(W2[(size_t)k * FOUT + n]) : (unsigned short)0;
  }
}

// edges + fused self loops; dst must be pre-initialized with bias
template <int F>
__global__ void k_scatter(const long long* __restrict__ ei,
                          const float* __restrict__ norm,
                          const float* __restrict__ dinv,
                          const float* __restrict__ src,
                          float* __restrict__ dst) {
  int i = blockIdx.x * blockDim.x + threadIdx.x;
  if (i >= (NEDGES + NNODES) * F) return;
  int e = i / F;
  int f = i - e * F;
  int r, c; float w;
  if (e < NEDGES) {
    r = (int)ei[e];
    c = (int)ei[NEDGES + e];
    w = norm[e];
  } else {
    r = c = e - NEDGES;
    float dv = dinv[r];
    w = dv * dv;
  }
  atomicAdd(&dst[(size_t)c * F + f], src[(size_t)r * F + f] * w);
}

// ---------------------------------------------------------------- GEMM 1
// H[N,128] = X16[N,512](bf16) @ W1T[128,512](bf16, n-major)
// block 256 thr, tile 64x128, K step 32, double-buffered async LDS staging.
// LDS row stride 40 ushorts = 80B: multiple of 16 (b128 aligned) and 20 banks
// -> 16 lanes' 4-bank spans tile all 64 banks (conflict-free).
#define A_LD 40
#define B_LD 40

__global__ __launch_bounds__(256) void k_gemm1(const unsigned short* __restrict__ X16,
                                               const unsigned short* __restrict__ W1T,
                                               float* __restrict__ H, int nrows) {
  __shared__ unsigned short As[2][64 * A_LD];    // A tile [m][k]
  __shared__ unsigned short Bst[2][128 * B_LD];  // B tile transposed [n][k]
  const int tid  = threadIdx.x;
  const int lane = tid & 31;
  const int wave = tid >> 5;
  const int wm   = (wave & 3) * 16;
  const int wn   = (wave >> 2) * 64;
  const int half = lane >> 4;
  const int ml   = lane & 15;
  const int m_block = blockIdx.x * 64;

  const unsigned asb = lds_off_of(&As[0][0]);
  const unsigned bsb = lds_off_of(&Bst[0][0]);

  // per-thread DMA assignments (fixed across the K loop)
  // A: 64 rows x 4 chunks of 16B = 256 transfers = 1/thread
  const int ar = tid >> 2, ac = tid & 3;
  int gra = m_block + ar; if (gra >= nrows) gra = nrows - 1;
  const char* asrc = (const char*)(X16 + (size_t)gra * FIN) + ac * 16;
  const unsigned arel = (unsigned)(ar * (A_LD * 2) + ac * 16);
  // B: 128 rows x 4 chunks = 512 transfers = 2/thread
  const int br0 = tid >> 2,         bc0 = tid & 3;
  const int br1 = (tid + 256) >> 2, bc1 = tid & 3;
  const char* bsrc0 = (const char*)(W1T + (size_t)br0 * FIN) + bc0 * 16;
  const char* bsrc1 = (const char*)(W1T + (size_t)br1 * FIN) + bc1 * 16;
  const unsigned brel0 = (unsigned)(br0 * (B_LD * 2) + bc0 * 16);
  const unsigned brel1 = (unsigned)(br1 * (B_LD * 2) + bc1 * 16);

  v8f acc[4] = {};

  // prologue: DMA tile 0 into buffer 0
  async_cp16(asb + arel, asrc);
  async_cp16(bsb + brel0, bsrc0);
  async_cp16(bsb + brel1, bsrc1);

  #pragma unroll
  for (int it = 0; it < FIN / 32; ++it) {
    const int k0  = it * 32;
    const int cur = it & 1;
    wait_async0();
    __syncthreads();
    if (k0 + 32 < FIN) {  // DMA next tile into the other buffer (static guard)
      const unsigned ab = asb + (cur ^ 1) * (64 * A_LD * 2);
      const unsigned bb = bsb + (cur ^ 1) * (128 * B_LD * 2);
      const size_t ko = (size_t)(k0 + 32) * 2;
      async_cp16(ab + arel, asrc + ko);
      async_cp16(bb + brel0, bsrc0 + ko);
      async_cp16(bb + brel1, bsrc1 + ko);
    }

    Frag16 a;
    {
      const unsigned short* ap = &As[cur][(wm + ml) * A_LD + 8 * half];
      a.q[0] = *(const uint4*)ap;
      a.q[1] = *(const uint4*)(ap + 16);
    }
    #pragma unroll
    for (int t = 0; t < 4; ++t) {
      Frag16 b;
      const unsigned short* bp = &Bst[cur][(wn + t * 16 + ml) * B_LD + 8 * half];
      b.q[0] = *(const uint4*)bp;
      b.q[1] = *(const uint4*)(bp + 16);
      acc[t] = __builtin_amdgcn_wmma_f32_16x16x32_bf16(
          false, a.v, false, b.v, (short)0, acc[t], false, false);
    }
  }

  #pragma unroll
  for (int t = 0; t < 4; ++t) {
    const int col = wn + t * 16 + ml;
    #pragma unroll
    for (int r = 0; r < 8; ++r) {
      int row = m_block + wm + r + 8 * half;
      if (row < nrows) H[(size_t)row * FHID + col] = acc[t][r];
    }
  }
}

// ---------------------------------------------------------------- GEMM 2
// H2[N,20] = H1[N,128](bf16) @ W2T[32,128](bf16, n-major, padded)
// block 256 thr, tile 128x32, K step 32, double-buffered async A staging,
// W2T DMA'd to LDS once. B2 k-stride 136 ushorts = 272B = 17x16 (conflict-free)
#define B2_LD 136

__global__ __launch_bounds__(256) void k_gemm2(const unsigned short* __restrict__ H1,
                                               const unsigned short* __restrict__ W2T,
                                               float* __restrict__ H2, int nrows) {
  __shared__ unsigned short As[2][128 * A_LD];
  __shared__ unsigned short Bs2[32 * B2_LD];
  const int tid  = threadIdx.x;
  const int lane = tid & 31;
  const int wave = tid >> 5;
  const int half = lane >> 4;
  const int ml   = lane & 15;
  const int m_block = blockIdx.x * 128;

  const unsigned asb = lds_off_of(&As[0][0]);
  const unsigned bsb = lds_off_of(&Bs2[0]);

  // stage whole W2T once: 32 rows x 16 chunks = 512 transfers = 2/thread
  #pragma unroll
  for (int i2 = 0; i2 < 2; ++i2) {
    int idx = tid + i2 * 256;
    int n = idx >> 4, ch = idx & 15;
    async_cp16(bsb + (unsigned)(n * (B2_LD * 2) + ch * 16),
               (const char*)(W2T + (size_t)n * FHID) + ch * 16);
  }

  // A: 128 rows x 4 chunks = 512 transfers = 2/thread
  const int r0 = tid >> 2,         c0 = tid & 3;
  const int r1 = (tid + 256) >> 2, c1 = tid & 3;
  int g0 = m_block + r0; if (g0 >= nrows) g0 = nrows - 1;
  int g1 = m_block + r1; if (g1 >= nrows) g1 = nrows - 1;
  const char* s0 = (const char*)(H1 + (size_t)g0 * FHID) + c0 * 16;
  const char* s1 = (const char*)(H1 + (size_t)g1 * FHID) + c1 * 16;
  const unsigned rel0 = (unsigned)(r0 * (A_LD * 2) + c0 * 16);
  const unsigned rel1 = (unsigned)(r1 * (A_LD * 2) + c1 * 16);

  v8f acc[2] = {};

  // prologue: A tile 0 -> buffer 0
  async_cp16(asb + rel0, s0);
  async_cp16(asb + rel1, s1);

  #pragma unroll
  for (int it = 0; it < FHID / 32; ++it) {
    const int k0  = it * 32;
    const int cur = it & 1;
    wait_async0();
    __syncthreads();
    if (k0 + 32 < FHID) {
      const unsigned ab = asb + (cur ^ 1) * (128 * A_LD * 2);
      const size_t ko = (size_t)(k0 + 32) * 2;
      async_cp16(ab + rel0, s0 + ko);
      async_cp16(ab + rel1, s1 + ko);
    }

    Frag16 a;
    {
      const unsigned short* ap = &As[cur][(wave * 16 + ml) * A_LD + 8 * half];
      a.q[0] = *(const uint4*)ap;
      a.q[1] = *(const uint4*)(ap + 16);
    }
    #pragma unroll
    for (int t = 0; t < 2; ++t) {
      Frag16 b;
      const unsigned short* bp = &Bs2[(t * 16 + ml) * B2_LD + k0 + 8 * half];
      b.q[0] = *(const uint4*)bp;
      b.q[1] = *(const uint4*)(bp + 16);
      acc[t] = __builtin_amdgcn_wmma_f32_16x16x32_bf16(
          false, a.v, false, b.v, (short)0, acc[t], false, false);
    }
  }

  #pragma unroll
  for (int t = 0; t < 2; ++t) {
    const int col = t * 16 + ml;
    if (col >= FOUT) continue;
    #pragma unroll
    for (int r = 0; r < 8; ++r) {
      int row = m_block + wave * 16 + r + 8 * half;
      if (row < nrows) H2[(size_t)row * FOUT + col] = acc[t][r];
    }
  }
}

// ---------------------------------------------------------------- launcher
static inline size_t align256(size_t x) { return (x + 255) & ~(size_t)255; }

extern "C" void kernel_launch(void* const* d_in, const int* in_sizes, int n_in,
                              void* d_out, int out_size, void* d_ws, size_t ws_size,
                              hipStream_t stream) {
  const float*     x  = (const float*)d_in[0];       // [50000,512]
  const long long* ei = (const long long*)d_in[1];   // [2,800000] int64
  const float*     ew = (const float*)d_in[2];       // [800000]
  const float*     W1 = (const float*)d_in[3];       // [512,128]
  const float*     b1 = (const float*)d_in[4];       // [128]
  const float*     W2 = (const float*)d_in[5];       // [128,20]
  const float*     b2 = (const float*)d_in[6];       // [20]
  float* out = (float*)d_out;                        // [50000,20]

  char* ws = (char*)d_ws;
  size_t off = 0;
  float*          deg  = (float*)(ws + off);          off += align256((size_t)NNODES * 4);
  float*          dinv = (float*)(ws + off);          off += align256((size_t)NNODES * 4);
  float*          nrm  = (float*)(ws + off);          off += align256((size_t)NEDGES * 4);
  unsigned short* x16  = (unsigned short*)(ws + off); off += align256((size_t)NNODES * FIN * 2);
  unsigned short* w1t  = (unsigned short*)(ws + off); off += align256((size_t)FHID * FIN * 2);
  unsigned short* w2t  = (unsigned short*)(ws + off); off += align256((size_t)32 * FHID * 2);
  float*          h    = (float*)(ws + off);          off += align256((size_t)NNODES * FHID * 4);
  float*          agg1 = (float*)(ws + off);          off += align256((size_t)NNODES * FHID * 4);
  unsigned short* h116 = (unsigned short*)(ws + off); off += align256((size_t)NNODES * FHID * 2);
  float*          h2   = h;  // h dead after scatter-1; reuse for layer-2 output

  const int B = 256;

  // normalization: deg = 1 + segment_sum(w, col); dinv = rsqrt(deg)
  k_fill<<<(NNODES + B - 1) / B, B, 0, stream>>>(deg, 1.0f, NNODES);
  k_deg_accum<<<(NEDGES + B - 1) / B, B, 0, stream>>>(ei, ew, deg);
  k_dinv<<<(NNODES + B - 1) / B, B, 0, stream>>>(deg, dinv);
  k_norm<<<(NEDGES + B - 1) / B, B, 0, stream>>>(ei, ew, dinv, nrm);

  // one-time bf16 conversions / transposes
  k_cvt_bf16<<<((NNODES * FIN / 4) + B - 1) / B, B, 0, stream>>>(x, x16, NNODES * FIN / 4);
  k_w1t<<<((FHID * FIN) + B - 1) / B, B, 0, stream>>>(W1, w1t);
  k_w2t<<<((32 * FHID) + B - 1) / B, B, 0, stream>>>(W2, w2t);

  // layer 1: h = x @ W1 (WMMA+async DMA), agg1 = b1 + scatter, relu->bf16
  k_gemm1<<<(NNODES + 63) / 64, B, 0, stream>>>(x16, w1t, h, NNODES);
  k_init_bias<<<((NNODES * FHID) + B - 1) / B, B, 0, stream>>>(agg1, b1, FHID,
                                                              NNODES * FHID);
  k_scatter<FHID><<<(((NEDGES + NNODES) * FHID) + B - 1) / B, B, 0, stream>>>(
      ei, nrm, dinv, h, agg1);
  k_relu_cvt<<<((NNODES * FHID / 4) + B - 1) / B, B, 0, stream>>>(
      agg1, h116, NNODES * FHID / 4);

  // layer 2: h2 = relu(agg1) @ W2 (WMMA+async DMA), out = b2 + scatter
  k_gemm2<<<(NNODES + 127) / 128, B, 0, stream>>>(h116, w2t, h2, NNODES);
  k_init_bias<<<((NNODES * FOUT) + B - 1) / B, B, 0, stream>>>(out, b2, FOUT,
                                                              NNODES * FOUT);
  k_scatter<FOUT><<<(((NEDGES + NNODES) * FOUT) + B - 1) / B, B, 0, stream>>>(
      ei, nrm, dinv, h2, out);
}
